// Joiner_15075335209363
// MI455X (gfx1250) — matrix operand order repaired and checked by reading, not compiled, not run on hardware
//
#include <hip/hip_runtime.h>
#include <hip/hip_bf16.h>

// Joiner (RNN-T): logits[n,t,u,v] = sum_j tanh(enc[n,t,j]+dec[n,u,j]) * W_out[v,j] + b_out[v]
// N=8 T=200 U=50 E=D=J=512 V=500.  Main GEMM: M=80000, K=512, Nv=500 (pad 512).
// Roofline: 160MB f32 output @ 23.3 TB/s ~= 6.9us store floor; 41 GFLOP of f16 WMMA is ~free.
// v3: 128-row x 512-col WG tiles (512 thr / 16 waves), wave tile 64x64.
//     W_out L2 traffic = 625 WGs x 512KB = 320MB (~parity with store floor).
//     Act tile 128x520 f16 (~130KB) in dynamic LDS (CDNA5: up to 320KB/WG).

typedef _Float16 h8  __attribute__((ext_vector_type(8)));
typedef _Float16 h16 __attribute__((ext_vector_type(16)));
typedef float    f8  __attribute__((ext_vector_type(8)));

union Frag16 { h16 v; h8 h[2]; };

// A-matrix fragment (16x32 f16, MxK). ISA 7.12.2:
// lanes 0-15: row M=lane, K chunks {k0..k0+7},{k0+16..k0+23}; lanes 16-31: row M=lane-16, +8.
__device__ __forceinline__ h16 load_a_frag(const _Float16* base, int ld, int row, int k0, int lane) {
  int koff = (lane & 16) ? 8 : 0;
  const _Float16* p = base + (size_t)row * ld + k0 + koff;
  Frag16 f;
  f.h[0] = *(const h8*)(p);
  f.h[1] = *(const h8*)(p + 16);
  return f.v;
}

// B-matrix fragment (32x16 f16, KxN), column n = lane&15 from row-major W[n, k].
// lanes 0-15 hold K chunks {k0..k0+7},{k0+16..k0+23}; lanes 16-31: +8 (matches A).
__device__ __forceinline__ h16 load_b_frag(const _Float16* W, int ld, int nrow, int k0, int lane) {
  int koff = (lane & 16) ? 8 : 0;
  const _Float16* p = W + (size_t)nrow * ld + k0 + koff;
  Frag16 f;
  f.h[0] = *(const h8*)(p);
  f.h[1] = *(const h8*)(p + 16);
  return f.v;
}

__global__ void cvt_f32_to_f16(const float* __restrict__ in, _Float16* __restrict__ out, int n) {
  int i = blockIdx.x * blockDim.x + threadIdx.x;
  if (i < n) out[i] = (_Float16)in[i];
}

// W_out (500x512) -> padded f16 (512x512), rows >= 500 zeroed
__global__ void cvt_pad_wout(const float* __restrict__ in, _Float16* __restrict__ out) {
  int i = blockIdx.x * blockDim.x + threadIdx.x;   // 0..262143
  int v = i >> 9, j = i & 511;
  out[i] = (v < 500) ? (_Float16)in[v * 512 + j] : (_Float16)0.f;
}

// out[m, j] = sum_k A16[m,k] * W16[j,k] + bias[j]; A: Mx512 f16, W: 512x512 f16, out f32.
// Grid: M/16 blocks of 256 threads (8 waves). Wave tile 16x64. (Projections are <3% of FLOPs.)
__global__ __launch_bounds__(256)
void proj_gemm_f16(const _Float16* __restrict__ A, const _Float16* __restrict__ W,
                   const float* __restrict__ bias, float* __restrict__ out) {
  const int m0   = blockIdx.x * 16;
  const int lane = threadIdx.x & 31;
  const int wv   = threadIdx.x >> 5;
  const int nb   = wv * 64;

  f8 acc[4] = {};
  const int arow = m0 + (lane & 15);
  for (int k0 = 0; k0 < 512; k0 += 32) {
    h16 a = load_a_frag(A, 512, arow, k0, lane);
#pragma unroll
    for (int i = 0; i < 4; ++i) {
      h16 b = load_b_frag(W, 512, nb + i * 16 + (lane & 15), k0, lane);
      acc[i] = __builtin_amdgcn_wmma_f32_16x16x32_f16(false, a, false, b, (short)0, acc[i], false, false);
    }
  }

  const int rowoff = (lane >> 4) << 3;   // 0 or 8
  const int col    = lane & 15;
#pragma unroll
  for (int i = 0; i < 4; ++i) {
    int j = nb + i * 16 + col;
    float bv = bias[j];
#pragma unroll
    for (int r = 0; r < 8; ++r)
      out[(size_t)(m0 + rowoff + r) * 512 + j] = acc[i][r] + bv;
  }
}

// Main fused kernel: per block (512 threads = 16 waves), one 128-row tile of (n,t,u).
// Stage f16(tanh(enc+dec)) 128x512 tile in dynamic LDS (130KB), then waves in a 2x8 grid
// each compute 64(M)x64(V): per k-step 4 A frags (ds) + 4 B frags (L2) -> 16 WMMAs.
#define LDS_LD  520   // 512 + 8 halfs pad to spread LDS banks
#define WG_ROWS 128
__global__ __launch_bounds__(512)
void joiner_kernel(const float* __restrict__ enc, const float* __restrict__ dec,
                   const _Float16* __restrict__ Wout, const float* __restrict__ bout,
                   float* __restrict__ out) {
  extern __shared__ _Float16 act[];          // WG_ROWS * LDS_LD halfs (~130KB)
  const int m0 = blockIdx.x * WG_ROWS;

  // build activation tile: WG_ROWS rows x 512 k, f16(tanh(enc_row + dec_row))
  for (int idx = threadIdx.x; idx < WG_ROWS * 512; idx += 512) {
    int row = idx >> 9;
    int k   = idx & 511;
    int m   = m0 + row;
    int n   = m / 10000;          // T*U = 10000
    int rem = m - n * 10000;
    int t   = rem / 50;
    int u   = rem - t * 50;
    float e = enc[(size_t)(n * 200 + t) * 512 + k];
    float d = dec[(size_t)(n * 50 + u) * 512 + k];
    act[row * LDS_LD + k] = (_Float16)tanhf(e + d);
  }
  __syncthreads();

  const int lane = threadIdx.x & 31;
  const int wv   = threadIdx.x >> 5;       // 0..15
  const int rg   = (wv >> 3) * 64;         // row group: 0 or 64
  const int vb   = (wv & 7) * 64;          // col group: 0..448

  f8 acc[4][4] = {};                       // [row_tile][col_tile], 128 VGPRs
  for (int k0 = 0; k0 < 512; k0 += 32) {
    h16 a[4];
#pragma unroll
    for (int r = 0; r < 4; ++r)
      a[r] = load_a_frag(act, LDS_LD, rg + r * 16 + (lane & 15), k0, lane);  // ds_load_b128 x2
#pragma unroll
    for (int i = 0; i < 4; ++i) {
      h16 b = load_b_frag(Wout, 512, vb + i * 16 + (lane & 15), k0, lane);   // L2-resident
#pragma unroll
      for (int r = 0; r < 4; ++r)
        acc[r][i] = __builtin_amdgcn_wmma_f32_16x16x32_f16(false, a[r], false, b, (short)0,
                                                           acc[r][i], false, false);
    }
  }

  const int rowoff = (lane >> 4) << 3;
  const int col    = lane & 15;
#pragma unroll
  for (int i = 0; i < 4; ++i) {
    int v = vb + i * 16 + col;
    if (v < 500) {
      float bv = bout[v];
#pragma unroll
      for (int r = 0; r < 4; ++r) {
        int mbase = m0 + rg + r * 16 + rowoff;
#pragma unroll
        for (int rr = 0; rr < 8; ++rr)
          out[(size_t)(mbase + rr) * 500 + v] = acc[r][i][rr] + bv;
      }
    }
  }
}

extern "C" void kernel_launch(void* const* d_in, const int* in_sizes, int n_in,
                              void* d_out, int out_size, void* d_ws, size_t ws_size,
                              hipStream_t stream) {
  const float* enc_in = (const float*)d_in[0];   // (8,200,512)  = 819200
  const float* dec_in = (const float*)d_in[1];   // (8,50,512)   = 204800
  const float* W_enc  = (const float*)d_in[2];   // (512,512)    = 262144
  const float* b_enc  = (const float*)d_in[3];   // 512
  const float* W_dec  = (const float*)d_in[4];   // (512,512)
  const float* b_dec  = (const float*)d_in[5];   // 512
  const float* W_out  = (const float*)d_in[6];   // (500,512)    = 256000
  const float* b_out  = (const float*)d_in[7];   // 500
  float* out = (float*)d_out;                    // (8,200,50,500) = 40,000,000

  char* ws = (char*)d_ws;
  size_t off = 0;
  auto alloc = [&](size_t bytes) -> void* {
    off = (off + 255) & ~(size_t)255;
    void* p = ws + off;
    off += bytes;
    return p;
  };

  _Float16* enc16  = (_Float16*)alloc((size_t)819200 * 2);
  _Float16* dec16  = (_Float16*)alloc((size_t)204800 * 2);
  _Float16* Wenc16 = (_Float16*)alloc((size_t)262144 * 2);
  _Float16* Wdec16 = (_Float16*)alloc((size_t)262144 * 2);
  _Float16* Wout16 = (_Float16*)alloc((size_t)262144 * 2);  // padded 512x512
  float*    encP   = (float*)alloc((size_t)1600 * 512 * 4); // enc projection (N*T, J)
  float*    decP   = (float*)alloc((size_t)400 * 512 * 4);  // dec projection (N*U, J)

  // f32 -> f16 staging
  cvt_f32_to_f16<<<(819200 + 255) / 256, 256, 0, stream>>>(enc_in, enc16, 819200);
  cvt_f32_to_f16<<<(204800 + 255) / 256, 256, 0, stream>>>(dec_in, dec16, 204800);
  cvt_f32_to_f16<<<(262144 + 255) / 256, 256, 0, stream>>>(W_enc, Wenc16, 262144);
  cvt_f32_to_f16<<<(262144 + 255) / 256, 256, 0, stream>>>(W_dec, Wdec16, 262144);
  cvt_pad_wout<<<262144 / 256, 256, 0, stream>>>(W_out, Wout16);

  // projections (WMMA): enc (1600x512), dec (400x512)
  proj_gemm_f16<<<100, 256, 0, stream>>>(enc16, Wenc16, b_enc, encP);
  proj_gemm_f16<<<25, 256, 0, stream>>>(dec16, Wdec16, b_dec, decP);

  // fused tanh + vocab GEMM: 80000/128 = 625 tiles, 512 threads, ~130KB dynamic LDS
  size_t lds_bytes = (size_t)WG_ROWS * LDS_LD * sizeof(_Float16);
  joiner_kernel<<<625, 512, lds_bytes, stream>>>(encP, decP, Wout16, b_out, out);
}
